// Global_block_10247791968285
// MI455X (gfx1250) — matrix operand order repaired and checked by reading, not compiled, not run on hardware
//
#include <hip/hip_runtime.h>

// ---------------------------------------------------------------------------
// Self-attention block (1x1-conv projections -> flash attention -> 1x1 conv
// -> BatchNorm(train) -> residual) for MI455X / gfx1250, wave32 + WMMA +
// async global->LDS double buffering in the attention kernel.
// b=8, C=256, D=128, N=64*64=4096.
// ---------------------------------------------------------------------------

typedef __attribute__((ext_vector_type(16))) _Float16 v16h;
typedef __attribute__((ext_vector_type(8)))  _Float16 v8h;
typedef __attribute__((ext_vector_type(8)))  float    v8f;
typedef __attribute__((ext_vector_type(4)))  float    v4f;

#define C_DIM 256
#define D_DIM 128
#define N_PIX 4096
#define B_SZ  8
#define KT    32   // key tile for flash attention (double buffered)

// ----- WMMA fragment loaders (gfx1250 wave32 layouts, 16-bit data) ----------
// A (16x32, MxK): lane l<16 holds row M=l, halves[0..7]=K 0..7, halves[8..15]=K 16..23
//                 lane l>=16 holds row M=l-16, K 8..15 and 24..31.
__device__ __forceinline__ v16h load_a16(const _Float16* base, int ld) {
  const int l  = threadIdx.x & 31;
  const _Float16* p = base + (l & 15) * ld + ((l >> 4) << 3);
  v8h lo = *(const v8h*)(p);
  v8h hi = *(const v8h*)(p + 16);
  v16h a;
#pragma unroll
  for (int i = 0; i < 8; ++i) { a[i] = lo[i]; a[i + 8] = hi[i]; }
  return a;
}

// B (32x16, KxN): lane l<16 holds column N=l, K 0..7 and 16..23; lane l>=16: K 8..15 and 24..31.
// Memory: column n contiguous over k at base + n*ld.
__device__ __forceinline__ v16h load_b16(const _Float16* base, int ld) {
  const int l  = threadIdx.x & 31;
  const _Float16* p = base + (l & 15) * ld + ((l >> 4) << 3);
  v8h lo = *(const v8h*)(p);
  v8h hi = *(const v8h*)(p + 16);
  v16h b;
#pragma unroll
  for (int i = 0; i < 8; ++i) { b[i] = lo[i]; b[i + 8] = hi[i]; }
  return b;
}

__device__ __forceinline__ v8f wmma_f16(v16h a, v16h b, v8f c) {
  return __builtin_amdgcn_wmma_f32_16x16x32_f16(false, a, false, b, (short)0, c,
                                                false, false);
}

// ----- CDNA5 async global->LDS (tracked by ASYNCcnt) ------------------------
__device__ __forceinline__ void async_load_b128(unsigned lds_off,
                                                const void* gaddr) {
  asm volatile("global_load_async_to_lds_b128 %0, %1, off"
               :: "v"(lds_off), "v"(gaddr) : "memory");
}
__device__ __forceinline__ void wait_async_le8() {
  asm volatile("s_wait_asynccnt 0x8" ::: "memory");
}
__device__ __forceinline__ void wait_async_0() {
  asm volatile("s_wait_asynccnt 0x0" ::: "memory");
}

// ----- Kernel 0: convert weights to f16, zero BN stat accumulators ----------
__global__ __launch_bounds__(256) void k_cvt(const float* __restrict__ Wt,
                                             const float* __restrict__ Wf,
                                             const float* __restrict__ Wg,
                                             const float* __restrict__ Wo,
                                             _Float16* __restrict__ WtH,
                                             _Float16* __restrict__ WfH,
                                             _Float16* __restrict__ WgH,
                                             _Float16* __restrict__ WoH,
                                             float* __restrict__ gsum,
                                             float* __restrict__ gsq) {
  int i = blockIdx.x * 256 + threadIdx.x;
  if (i < D_DIM * C_DIM) {
    WtH[i] = (_Float16)Wt[i];
    WfH[i] = (_Float16)Wf[i];
    WgH[i] = (_Float16)Wg[i];
    WoH[i] = (_Float16)Wo[i];
  }
  if (i < C_DIM) { gsum[i] = 0.f; gsq[i] = 0.f; }
}

// ----- Kernel 1: projections t,f,g (1x1 conv as GEMM) -----------------------
// out tT,fT: [b][N][D] f16 ; gO: [b][D][N] f16
__global__ __launch_bounds__(256) void k_proj(
    const float* __restrict__ x,            // [b][C][N]
    const _Float16* __restrict__ WtH, const _Float16* __restrict__ WfH,
    const _Float16* __restrict__ WgH,
    const float* __restrict__ bt, const float* __restrict__ bf,
    const float* __restrict__ bg,
    _Float16* __restrict__ tT, _Float16* __restrict__ fT,
    _Float16* __restrict__ gO) {
  const int b  = blockIdx.y;
  const int n0 = blockIdx.x * 64;
  __shared__ _Float16 sx[64][C_DIM + 8];    // x^T tile [pixel][channel]

  const int wave = threadIdx.x >> 5;        // 0..7 -> d-subtile
  const int d0   = wave * 16;
  __builtin_prefetch(WtH + (size_t)d0 * C_DIM, 0, 3);
  __builtin_prefetch(WfH + (size_t)d0 * C_DIM, 0, 3);
  __builtin_prefetch(WgH + (size_t)d0 * C_DIM, 0, 3);

  const float* xb = x + (size_t)b * C_DIM * N_PIX;
  for (int idx = threadIdx.x; idx < C_DIM * 64; idx += 256) {
    int c = idx >> 6, m = idx & 63;
    sx[m][c] = (_Float16)xb[(size_t)c * N_PIX + n0 + m];
  }
  __syncthreads();

  const int lane = threadIdx.x & 31;
  const int ccol = lane & 15;
  const int mbase = (lane >> 4) * 8;
  const float biast = bt[d0 + ccol];
  const float biasf = bf[d0 + ccol];
  const float biasg = bg[d0 + ccol];
  const size_t bnd = (size_t)b * N_PIX * D_DIM;

  for (int sub = 0; sub < 4; ++sub) {
    v8f at = {}, af = {}, ag = {};
#pragma unroll
    for (int k = 0; k < C_DIM; k += 32) {
      v16h a  = load_a16(&sx[sub * 16][k], C_DIM + 8);
      v16h b0 = load_b16(WtH + (size_t)d0 * C_DIM + k, C_DIM);
      v16h b1 = load_b16(WfH + (size_t)d0 * C_DIM + k, C_DIM);
      v16h b2 = load_b16(WgH + (size_t)d0 * C_DIM + k, C_DIM);
      at = wmma_f16(a, b0, at);
      af = wmma_f16(a, b1, af);
      ag = wmma_f16(a, b2, ag);
    }
#pragma unroll
    for (int r = 0; r < 8; ++r) {
      int n = n0 + sub * 16 + mbase + r;
      tT[bnd + (size_t)n * D_DIM + d0 + ccol] = (_Float16)(at[r] + biast);
      fT[bnd + (size_t)n * D_DIM + d0 + ccol] = (_Float16)(af[r] + biasf);
    }
    v8h gv;
#pragma unroll
    for (int r = 0; r < 8; ++r) gv[r] = (_Float16)(ag[r] + biasg);
    *(v8h*)&gO[(size_t)b * D_DIM * N_PIX + (size_t)(d0 + ccol) * N_PIX +
               n0 + sub * 16 + mbase] = gv;
  }
}

// ----- Kernel 2: flash attention  y^T = softmax(t^T f) g^T ------------------
// tT,fT: [b][N][D] f16 ; gI: [b][D][N] f16 ; yT out: [b][N][D] f16
// 4 waves x 16 queries; key tiles of 32, double-buffered via async LDS loads.
__global__ __launch_bounds__(128) void k_attn(const _Float16* __restrict__ tT,
                                              const _Float16* __restrict__ fT,
                                              const _Float16* __restrict__ gI,
                                              _Float16* __restrict__ yT) {
  const int b  = blockIdx.y;
  const int q0 = blockIdx.x * 64;
  __shared__ _Float16 st[64][D_DIM];        // query tile (16KB)
  __shared__ _Float16 sf[2][KT][D_DIM];     // key tiles f^T (2x8KB)
  __shared__ _Float16 sg[2][D_DIM][KT];     // value tiles g (2x8KB)
  __shared__ _Float16 sp[4][16][KT + 8];    // per-wave P tile (5KB)

  const size_t bnd = (size_t)b * N_PIX * D_DIM;
  const size_t bgd = (size_t)b * D_DIM * N_PIX;
  const int tid = threadIdx.x;

  // issue one 32-key f/g tile as async global->LDS copies (8 ops/thread)
  auto issue_tile = [&](int ktile, int buf) {
    const int m0 = ktile * KT;
#pragma unroll
    for (int k = 0; k < 4; ++k) {
      int idx = tid + k * 128;            // 0..511
      int m = idx >> 4, v = idx & 15;     // sf: 32 rows x 16 16B-chunks
      async_load_b128((unsigned)(uintptr_t)&sf[buf][m][v * 8],
                      &fT[bnd + (size_t)(m0 + m) * D_DIM + v * 8]);
    }
#pragma unroll
    for (int k = 0; k < 4; ++k) {
      int idx = tid + k * 128;
      int d = idx >> 2, v = idx & 3;      // sg: 128 rows x 4 16B-chunks
      async_load_b128((unsigned)(uintptr_t)&sg[buf][d][v * 8],
                      &gI[bgd + (size_t)d * N_PIX + m0 + v * 8]);
    }
  };

  issue_tile(0, 0);                       // prefetch first tile

  // query tile (one-time, plain loads)
  for (int idx = tid; idx < 64 * 16; idx += 128) {
    int m = idx >> 4, v = idx & 15;
    *(v8h*)&st[m][v * 8] = *(const v8h*)&tT[bnd + (size_t)(q0 + m) * D_DIM + v * 8];
  }

  const int wave = tid >> 5;              // 0..3 -> 16-query subtile
  const int lane = tid & 31;
  const int mbase = (lane >> 4) * 8;
  const int ccol  = lane & 15;

  v8f acc[8];
#pragma unroll
  for (int dt = 0; dt < 8; ++dt) acc[dt] = (v8f){};
  float rmax[8], rsum[8];
#pragma unroll
  for (int r = 0; r < 8; ++r) { rmax[r] = -3.0e38f; rsum[r] = 0.f; }

  const int niter = N_PIX / KT;           // 128
  for (int i = 0; i < niter; ++i) {
    const int cur = i & 1;
    if (i + 1 < niter) {
      issue_tile(i + 1, cur ^ 1);
      wait_async_le8();                   // tile i complete, tile i+1 in flight
    } else {
      wait_async_0();
    }
    __syncthreads();

    // scores S = t^T f : 2 key subtiles, K = D = 128
    v8f s2[2];
#pragma unroll
    for (int j = 0; j < 2; ++j) {
      v8f sc = {};
#pragma unroll
      for (int k = 0; k < D_DIM; k += 32) {
        v16h a  = load_a16(&st[wave * 16][k], D_DIM);
        v16h bb = load_b16(&sf[cur][j * 16][k], D_DIM);
        sc = wmma_f16(a, bb, sc);
      }
      s2[j] = sc;
    }

    // online softmax: row = query lives in a 16-lane half at fixed vgpr r
    float scale[8];
#pragma unroll
    for (int r = 0; r < 8; ++r) {
      float m = fmaxf(s2[0][r], s2[1][r]);
      m = fmaxf(m, __shfl_xor(m, 1, 32));
      m = fmaxf(m, __shfl_xor(m, 2, 32));
      m = fmaxf(m, __shfl_xor(m, 4, 32));
      m = fmaxf(m, __shfl_xor(m, 8, 32));
      float nm = fmaxf(rmax[r], m);
      scale[r] = __expf(rmax[r] - nm);
      rmax[r]  = nm;
    }
#pragma unroll
    for (int j = 0; j < 2; ++j) {
#pragma unroll
      for (int r = 0; r < 8; ++r) {
        float p = __expf(s2[j][r] - rmax[r]);
        s2[j][r] = p;
        sp[wave][mbase + r][j * 16 + ccol] = (_Float16)p;   // D-layout -> LDS
      }
    }
#pragma unroll
    for (int r = 0; r < 8; ++r) {
      float ps = s2[0][r] + s2[1][r];
      ps += __shfl_xor(ps, 1, 32);
      ps += __shfl_xor(ps, 2, 32);
      ps += __shfl_xor(ps, 4, 32);
      ps += __shfl_xor(ps, 8, 32);
      rsum[r] = rsum[r] * scale[r] + ps;
    }

    // Y = Y*scale + P (16x32) * g^T (32x128)
    v16h pa = load_a16(&sp[wave][0][0], KT + 8);
#pragma unroll
    for (int dt = 0; dt < 8; ++dt) {
      v8f a = acc[dt];
#pragma unroll
      for (int r = 0; r < 8; ++r) a[r] *= scale[r];
      v16h bb = load_b16(&sg[cur][dt * 16][0], KT);
      acc[dt] = wmma_f16(pa, bb, a);
    }
    __syncthreads();   // all waves done with buf cur before it is refilled
  }

  // finalize: divide by row sums, store y^T [b][N][D]
#pragma unroll
  for (int dt = 0; dt < 8; ++dt) {
#pragma unroll
    for (int r = 0; r < 8; ++r) {
      int q = q0 + wave * 16 + mbase + r;
      yT[bnd + (size_t)q * D_DIM + dt * 16 + ccol] =
          (_Float16)(acc[dt][r] / rsum[r]);
    }
  }
}

// ----- Kernel 3: z = Wo y + bo, store z, accumulate BN stats ----------------
__global__ __launch_bounds__(256) void k_outconv(
    const _Float16* __restrict__ yT, const _Float16* __restrict__ WoH,
    const float* __restrict__ bo, float* __restrict__ z,
    float* __restrict__ gsum, float* __restrict__ gsq) {
  const int b  = blockIdx.y;
  const int n0 = blockIdx.x * 64;
  __shared__ _Float16 sy[64][D_DIM];        // 16KB
  __shared__ float ssum[C_DIM], ssq[C_DIM];
  if (threadIdx.x < C_DIM) { ssum[threadIdx.x] = 0.f; ssq[threadIdx.x] = 0.f; }
  __builtin_prefetch(WoH + (size_t)(threadIdx.x >> 5) * 2 * 16 * D_DIM, 0, 3);

  const size_t bnd = (size_t)b * N_PIX * D_DIM;
  for (int idx = threadIdx.x; idx < 64 * 16; idx += 256) {
    int m = idx >> 4, v = idx & 15;
    *(v8h*)&sy[m][v * 8] = *(const v8h*)&yT[bnd + (size_t)(n0 + m) * D_DIM + v * 8];
  }
  __syncthreads();

  const int wave = threadIdx.x >> 5;
  const int lane = threadIdx.x & 31;
  const int mbase = (lane >> 4) * 8;
  const int ccol  = lane & 15;

  for (int ci = 0; ci < 2; ++ci) {
    const int c0 = (wave * 2 + ci) * 16;
    const int c  = c0 + ccol;
    const float bias = bo[c];
    for (int sub = 0; sub < 4; ++sub) {
      v8f accz = {};
#pragma unroll
      for (int k = 0; k < D_DIM; k += 32) {
        v16h a  = load_a16(&sy[sub * 16][k], D_DIM);
        v16h bb = load_b16(WoH + (size_t)c0 * D_DIM + k, D_DIM);
        accz = wmma_f16(a, bb, accz);
      }
      const int n = n0 + sub * 16 + mbase;
      float sum8 = 0.f, sq8 = 0.f;
      v4f lo, hi;
#pragma unroll
      for (int r = 0; r < 8; ++r) {
        float v = accz[r] + bias;
        if (r < 4) lo[r] = v; else hi[r - 4] = v;
        sum8 += v;
        sq8  += v * v;
      }
      float* zp = z + (size_t)b * C_DIM * N_PIX + (size_t)c * N_PIX + n;
      *(v4f*)zp       = lo;
      *(v4f*)(zp + 4) = hi;
      atomicAdd(&ssum[c], sum8);
      atomicAdd(&ssq[c],  sq8);
    }
  }
  __syncthreads();
  if (threadIdx.x < C_DIM) {
    atomicAdd(&gsum[threadIdx.x], ssum[threadIdx.x]);
    atomicAdd(&gsq[threadIdx.x],  ssq[threadIdx.x]);
  }
}

// ----- Kernel 4: BatchNorm (batch stats) + affine + residual ----------------
__global__ __launch_bounds__(256) void k_bnres(const float* __restrict__ x,
                                               float* __restrict__ out,
                                               const float* __restrict__ gsum,
                                               const float* __restrict__ gsq,
                                               const float* __restrict__ gamma,
                                               const float* __restrict__ beta) {
  const size_t i = (size_t)blockIdx.x * 256 + threadIdx.x;
  const int c = (int)((i / N_PIX) % C_DIM);
  const float cnt  = (float)(B_SZ * N_PIX);
  const float mean = gsum[c] / cnt;
  const float var  = gsq[c] / cnt - mean * mean;
  const float rstd = rsqrtf(var + 1e-5f);
  const float zv   = out[i];
  out[i] = (zv - mean) * rstd * gamma[c] + beta[c] + x[i];
}

// ---------------------------------------------------------------------------
extern "C" void kernel_launch(void* const* d_in, const int* in_sizes, int n_in,
                              void* d_out, int out_size, void* d_ws,
                              size_t ws_size, hipStream_t stream) {
  const float* x     = (const float*)d_in[0];
  const float* Wt    = (const float*)d_in[1];
  const float* bt    = (const float*)d_in[2];
  const float* Wf    = (const float*)d_in[3];
  const float* bf    = (const float*)d_in[4];
  const float* Wg    = (const float*)d_in[5];
  const float* bg    = (const float*)d_in[6];
  const float* Wo    = (const float*)d_in[7];
  const float* bo    = (const float*)d_in[8];
  const float* gamma = (const float*)d_in[9];
  const float* beta  = (const float*)d_in[10];
  float* out = (float*)d_out;

  // workspace carve-up (ws assumed 256B aligned)
  _Float16* WtH = (_Float16*)d_ws;
  _Float16* WfH = WtH + (size_t)D_DIM * C_DIM;
  _Float16* WgH = WfH + (size_t)D_DIM * C_DIM;
  _Float16* WoH = WgH + (size_t)D_DIM * C_DIM;
  _Float16* tT  = WoH + (size_t)C_DIM * D_DIM;
  _Float16* fT  = tT + (size_t)B_SZ * N_PIX * D_DIM;
  _Float16* gP  = fT + (size_t)B_SZ * N_PIX * D_DIM;
  _Float16* yT  = gP + (size_t)B_SZ * N_PIX * D_DIM;
  float* gsum   = (float*)(yT + (size_t)B_SZ * N_PIX * D_DIM);
  float* gsq    = gsum + C_DIM;

  k_cvt<<<(D_DIM * C_DIM + 255) / 256, 256, 0, stream>>>(
      Wt, Wf, Wg, Wo, WtH, WfH, WgH, WoH, gsum, gsq);

  dim3 gproj(N_PIX / 64, B_SZ);
  k_proj<<<gproj, 256, 0, stream>>>(x, WtH, WfH, WgH, bt, bf, bg, tT, fT, gP);

  dim3 gattn(N_PIX / 64, B_SZ);
  k_attn<<<gattn, 128, 0, stream>>>(tT, fT, gP, yT);

  dim3 gout(N_PIX / 64, B_SZ);
  k_outconv<<<gout, 256, 0, stream>>>(yT, WoH, bo, out, gsum, gsq);

  const size_t total = (size_t)B_SZ * C_DIM * N_PIX;
  k_bnres<<<(unsigned)(total / 256), 256, 0, stream>>>(x, out, gsum, gsq,
                                                       gamma, beta);
}
